// StratifiedTransformerBlock_420906795536
// MI455X (gfx1250) — matrix operand order repaired and checked by reading, not compiled, not run on hardware
//
#include <hip/hip_runtime.h>
#include <hip/hip_bf16.h>
#include <math.h>

typedef __attribute__((ext_vector_type(16))) __bf16 v16bf;
typedef __attribute__((ext_vector_type(8)))  __bf16 bf16x8;
typedef __attribute__((ext_vector_type(8)))  float  v8f;
typedef __attribute__((ext_vector_type(2)))  float  v2f;

namespace {
constexpr int NT = 32768;    // tokens
constexpr int CH = 96;       // channels
constexpr int NH = 6;        // heads
constexpr int HD = 16;       // head dim
constexpr int MW = 32;       // window size
constexpr int NW = NT / MW;  // 1024 windows
constexpr float QK_SCALE = 0.25f;  // HD^-0.5
}

// ---------------------------------------------------------------------------
// Weight prep: f32 [K][N] row-major  ->  bf16 [N][K] (K contiguous per column)
// ---------------------------------------------------------------------------
__global__ __launch_bounds__(256)
void wprep_kernel(const float* __restrict__ Wsrc, __bf16* __restrict__ BT,
                  int K, int N) {
  int t = blockIdx.x * 256 + threadIdx.x;
  if (t >= K * N) return;
  int n = t / K, k = t % K;
  BT[t] = (__bf16)Wsrc[k * N + n];
}

// ---------------------------------------------------------------------------
// Mega-fused kernel 1: per 32-token window (1024 blocks x 8 wave32)
//   x rows -> LDS -> LN1 -> QKV GEMM (bf16 WMMA) -> q/k/v in LDS
//   -> window attention (f32 WMMA + bias tables + softmax + t_v gather)
//   -> proj GEMM (bf16 WMMA) + residual -> y (only HBM intermediate)
// ---------------------------------------------------------------------------
__global__ __launch_bounds__(256)
void block_attn_kernel(const float* __restrict__ x, const float* __restrict__ coords,
                       const float* __restrict__ gamma1, const float* __restrict__ beta1,
                       const __bf16* __restrict__ wqkvT, const float* __restrict__ b_qkv,
                       const float* __restrict__ t_q, const float* __restrict__ t_k,
                       const float* __restrict__ t_v,
                       const __bf16* __restrict__ wprojT, const float* __restrict__ b_proj,
                       float* __restrict__ y) {
  extern __shared__ char smem_raw[];
  float*  x_s   = (float*)smem_raw;            // 32 x 96 (residual source)
  float*  q_s   = x_s  + MW * CH;              // 32 x 96 (pre-scaled q)
  float*  k_s   = q_s  + MW * CH;
  float*  v_s   = k_s  + MW * CH;
  float*  att   = v_s  + MW * CH;              // 6 x 32 x 32
  float*  out_s = att  + NH * MW * MW;         // 32 x 96
  float*  tv_s  = out_s + MW * CH;             // 16*3*6*16 = 4608 (t_v cache)
  float*  c_s   = tv_s + 4608;                 // 32 x 3
  __bf16* xn_s  = (__bf16*)(c_s + 128);        // 32 x 96 (LN1 out, WMMA A)
  __bf16* ob_s  = xn_s + MW * CH;              // 32 x 96 (attn out, WMMA A)
  __bf16* qt_s  = ob_s + MW * CH;              // [i][h][a][bin] = 9216
  __bf16* kt_s  = qt_s + MW * NH * 3 * 16;     // 9216
  unsigned short* idx_s = (unsigned short*)(kt_s + MW * NH * 3 * 16); // 1024

  const int tid  = threadIdx.x;
  const int lane = tid & 31, warp = tid >> 5;
  const int r    = lane & 15, hf = lane >> 4;
  const int w    = blockIdx.x;

  // ---- P0: stage x rows, coords, t_v table
  for (int t = tid; t < MW * CH; t += 256) x_s[t] = x[(size_t)w * MW * CH + t];
  for (int t = tid; t < MW * 3; t += 256)  c_s[t] = coords[(size_t)w * MW * 3 + t];
  for (int t = tid; t < 4608; t += 256)    tv_s[t] = t_v[t];
  __syncthreads();

  // ---- P1: LayerNorm1 (one wave32 per row, 4 rows per wave)
  for (int rr = 0; rr < 4; ++rr) {
    int row = warp * 4 + rr;
    float v0 = x_s[row * CH + lane];
    float v1 = x_s[row * CH + lane + 32];
    float v2 = x_s[row * CH + lane + 64];
    float s = v0 + v1 + v2;
    for (int o = 16; o; o >>= 1) s += __shfl_xor(s, o, 32);
    float mean = s * (1.0f / 96.0f);
    float d0 = v0 - mean, d1 = v1 - mean, d2 = v2 - mean;
    float q = d0 * d0 + d1 * d1 + d2 * d2;
    for (int o = 16; o; o >>= 1) q += __shfl_xor(q, o, 32);
    float inv = rsqrtf(q * (1.0f / 96.0f) + 1e-5f);
    xn_s[row * CH + lane]      = (__bf16)(d0 * inv * gamma1[lane]      + beta1[lane]);
    xn_s[row * CH + lane + 32] = (__bf16)(d1 * inv * gamma1[lane + 32] + beta1[lane + 32]);
    xn_s[row * CH + lane + 64] = (__bf16)(d2 * inv * gamma1[lane + 64] + beta1[lane + 64]);
  }
  __syncthreads();

  // ---- P2: QKV GEMM (32x96 @ 96x288), bf16 WMMA, results into q/k/v LDS
  for (int wt = warp; wt < 2 * 18; wt += 8) {
    int tM = wt / 18, tN = wt % 18;
    const __bf16* aP = xn_s + (tM * 16 + r) * CH + hf * 8;
    const __bf16* bP = wqkvT + (size_t)(tN * 16 + r) * CH + hf * 8;
    v8f acc = {};
    for (int k = 0; k < CH; k += 32) {
      bf16x8 a0 = *(const bf16x8*)(aP + k);
      bf16x8 a1 = *(const bf16x8*)(aP + k + 16);
      bf16x8 b0 = *(const bf16x8*)(bP + k);
      bf16x8 b1 = *(const bf16x8*)(bP + k + 16);
      v16bf av = __builtin_shufflevector(a0, a1, 0,1,2,3,4,5,6,7,8,9,10,11,12,13,14,15);
      v16bf bv = __builtin_shufflevector(b0, b1, 0,1,2,3,4,5,6,7,8,9,10,11,12,13,14,15);
      acc = __builtin_amdgcn_wmma_f32_16x16x32_bf16(false, av, false, bv,
                                                    (short)0, acc, false, false);
    }
    int col = tN * 16 + r, rbase = tM * 16 + hf * 8;
    float bc = b_qkv[col];
    for (int rr = 0; rr < 8; ++rr) {
      int row = rbase + rr;
      float vv = acc[rr] + bc;
      if (col < CH)          q_s[row * CH + col]          = vv * QK_SCALE;
      else if (col < 2 * CH) k_s[row * CH + (col - CH)]   = vv;
      else                   v_s[row * CH + (col - 2*CH)] = vv;
    }
  }
  __syncthreads();

  // ---- P3a: bin indices, 3 x 4-bit packed per (i,j)
  for (int p = tid; p < MW * MW; p += 256) {
    int i = p >> 5, j = p & 31;
    unsigned pk = 0;
    for (int a = 0; a < 3; ++a) {
      float rel = c_s[i * 3 + a] - c_s[j * 3 + a];
      int bidx = (int)floorf((rel * 0.25f + 1.0f) * 8.0f);
      bidx = bidx < 0 ? 0 : (bidx > 15 ? 15 : bidx);
      pk |= (unsigned)bidx << (4 * a);
    }
    idx_s[p] = (unsigned short)pk;
  }

  // ---- P3b: qt/kt bias tables (bf16 in LDS)
  for (int e = tid; e < MW * NH * 3 * 16; e += 256) {
    int bidx = e & 15, a = (e >> 4) % 3, h = (e / 48) % 6, i = e / 288;
    const float* tq = t_q + (((bidx * 3 + a) * 6 + h) << 4);
    const float* tk = t_k + (((bidx * 3 + a) * 6 + h) << 4);
    const float* qr = q_s + i * CH + h * HD;
    const float* kr = k_s + i * CH + h * HD;
    float sq = 0.f, sk = 0.f;
    for (int d = 0; d < HD; ++d) { sq += qr[d] * tq[d]; sk += kr[d] * tk[d]; }
    qt_s[e] = (__bf16)sq;
    kt_s[e] = (__bf16)(sk * QK_SCALE);
  }

  // ---- P3c: att[h][i][j] = q . k via f32 WMMA (24 16x16 tiles, K=16)
  for (int wt = warp; wt < 24; wt += 8) {
    int h = wt >> 2, it = (wt >> 1) & 1, jt = wt & 1;
    v8f acc = {};
    for (int kk = 0; kk < 4; ++kk) {
      int dp = kk * 4 + 2 * hf;
      v2f av, bv;
      av.x = q_s[(it * 16 + r) * CH + h * HD + dp];
      av.y = q_s[(it * 16 + r) * CH + h * HD + dp + 1];
      bv.x = k_s[(jt * 16 + r) * CH + h * HD + dp];
      bv.y = k_s[(jt * 16 + r) * CH + h * HD + dp + 1];
      acc = __builtin_amdgcn_wmma_f32_16x16x4_f32(false, av, false, bv,
                                                  (short)0, acc, false, false);
    }
    for (int rr = 0; rr < 8; ++rr) {
      int row = it * 16 + rr + 8 * hf;
      att[(h * MW + row) * MW + jt * 16 + r] = acc[rr];
    }
  }
  __syncthreads();

  // ---- P4: add relative-position bias
  for (int p = tid; p < NH * MW * MW; p += 256) {
    int j = p & 31, i = (p >> 5) & 31, h = p >> 10;
    unsigned pk = idx_s[i * 32 + j];
    float bias = 0.f;
    for (int a = 0; a < 3; ++a) {
      int bidx = (pk >> (4 * a)) & 15;
      bias += (float)qt_s[((i * 6 + h) * 3 + a) * 16 + bidx];
      bias += (float)kt_s[((j * 6 + h) * 3 + a) * 16 + bidx];
    }
    att[p] += bias;
  }
  __syncthreads();

  // ---- P5: softmax over j (192 rows)
  if (tid < NH * MW) {
    float* row = att + tid * MW;
    float mx = row[0];
    for (int j = 1; j < MW; ++j) mx = fmaxf(mx, row[j]);
    float s = 0.f;
    for (int j = 0; j < MW; ++j) { float e = __expf(row[j] - mx); row[j] = e; s += e; }
    float inv = 1.0f / s;
    for (int j = 0; j < MW; ++j) row[j] *= inv;
  }
  __syncthreads();

  // ---- P6a: out = softmax @ v via f32 WMMA (12 tiles, K=32)
  for (int wt = warp; wt < 12; wt += 8) {
    int h = wt >> 1, it = wt & 1;
    v8f acc = {};
    for (int kk = 0; kk < 8; ++kk) {
      int kj = kk * 4 + 2 * hf;
      v2f av, bv;
      av.x = att[(h * MW + it * 16 + r) * MW + kj];
      av.y = att[(h * MW + it * 16 + r) * MW + kj + 1];
      bv.x = v_s[kj * CH + h * HD + r];
      bv.y = v_s[(kj + 1) * CH + h * HD + r];
      acc = __builtin_amdgcn_wmma_f32_16x16x4_f32(false, av, false, bv,
                                                  (short)0, acc, false, false);
    }
    for (int rr = 0; rr < 8; ++rr) {
      int row = it * 16 + rr + 8 * hf;
      out_s[row * CH + h * HD + r] = acc[rr];
    }
  }
  __syncthreads();

  // ---- P6b: out += sum_j a[h,i,j] * sum_a t_v[idx(i,j,a), a, h, :]
  for (int e = tid; e < MW * CH; e += 256) {
    int d = e & 15, h = (e >> 4) % 6, i = e / CH;
    const float* arow = att + (h * MW + i) * MW;
    const unsigned short* irow = idx_s + i * MW;
    float acc2 = 0.f;
    for (int j = 0; j < MW; ++j) {
      unsigned pk = irow[j];
      float tvsum = tv_s[(((( pk       & 15) * 3 + 0) * 6 + h) << 4) + d]
                  + tv_s[(((((pk >> 4) & 15) * 3 + 1) * 6 + h) << 4) + d]
                  + tv_s[(((((pk >> 8) & 15) * 3 + 2) * 6 + h) << 4) + d];
      acc2 += arow[j] * tvsum;
    }
    out_s[i * CH + h * HD + d] += acc2;
  }
  __syncthreads();

  // ---- P7: bf16 copy of attention output (A operand of proj GEMM)
  for (int t = tid; t < MW * CH; t += 256) ob_s[t] = (__bf16)out_s[t];
  __syncthreads();

  // ---- P8: proj GEMM (32x96 @ 96x96) + residual -> y
  for (int wt = warp; wt < 12; wt += 8) {
    int tM = wt / 6, tN = wt % 6;
    const __bf16* aP = ob_s + (tM * 16 + r) * CH + hf * 8;
    const __bf16* bP = wprojT + (size_t)(tN * 16 + r) * CH + hf * 8;
    v8f acc = {};
    for (int k = 0; k < CH; k += 32) {
      bf16x8 a0 = *(const bf16x8*)(aP + k);
      bf16x8 a1 = *(const bf16x8*)(aP + k + 16);
      bf16x8 b0 = *(const bf16x8*)(bP + k);
      bf16x8 b1 = *(const bf16x8*)(bP + k + 16);
      v16bf av = __builtin_shufflevector(a0, a1, 0,1,2,3,4,5,6,7,8,9,10,11,12,13,14,15);
      v16bf bv = __builtin_shufflevector(b0, b1, 0,1,2,3,4,5,6,7,8,9,10,11,12,13,14,15);
      acc = __builtin_amdgcn_wmma_f32_16x16x32_bf16(false, av, false, bv,
                                                    (short)0, acc, false, false);
    }
    int col = tN * 16 + r, rbase = tM * 16 + hf * 8;
    float bc = b_proj[col];
    for (int rr = 0; rr < 8; ++rr) {
      int row = rbase + rr;
      y[((size_t)w * MW + row) * CH + col] = acc[rr] + bc + x_s[row * CH + col];
    }
  }
}

// ---------------------------------------------------------------------------
// Mega-fused kernel 2: per 32-row chunk
//   y rows -> LDS -> LN2 -> MLP1+GELU (bf16 WMMA, LDS) -> MLP2 (K=384) + resid
// ---------------------------------------------------------------------------
__global__ __launch_bounds__(256)
void block_mlp_kernel(const float* __restrict__ y,
                      const float* __restrict__ gamma2, const float* __restrict__ beta2,
                      const __bf16* __restrict__ wmlp1T, const float* __restrict__ b_mlp1,
                      const __bf16* __restrict__ wmlp2T, const float* __restrict__ b_mlp2,
                      float* __restrict__ out) {
  extern __shared__ char smem_raw[];
  float*  y_s = (float*)smem_raw;          // 32 x 96
  __bf16* h_s = (__bf16*)(y_s + MW * CH);  // 32 x 96 (LN2 out)
  __bf16* g_s = h_s + MW * CH;             // 32 x 384 (gelu out)

  const int tid  = threadIdx.x;
  const int lane = tid & 31, warp = tid >> 5;
  const int r    = lane & 15, hf = lane >> 4;
  const int w    = blockIdx.x;

  // ---- stage y rows
  for (int t = tid; t < MW * CH; t += 256) y_s[t] = y[(size_t)w * MW * CH + t];
  __syncthreads();

  // ---- LayerNorm2
  for (int rr = 0; rr < 4; ++rr) {
    int row = warp * 4 + rr;
    float v0 = y_s[row * CH + lane];
    float v1 = y_s[row * CH + lane + 32];
    float v2 = y_s[row * CH + lane + 64];
    float s = v0 + v1 + v2;
    for (int o = 16; o; o >>= 1) s += __shfl_xor(s, o, 32);
    float mean = s * (1.0f / 96.0f);
    float d0 = v0 - mean, d1 = v1 - mean, d2 = v2 - mean;
    float q = d0 * d0 + d1 * d1 + d2 * d2;
    for (int o = 16; o; o >>= 1) q += __shfl_xor(q, o, 32);
    float inv = rsqrtf(q * (1.0f / 96.0f) + 1e-5f);
    h_s[row * CH + lane]      = (__bf16)(d0 * inv * gamma2[lane]      + beta2[lane]);
    h_s[row * CH + lane + 32] = (__bf16)(d1 * inv * gamma2[lane + 32] + beta2[lane + 32]);
    h_s[row * CH + lane + 64] = (__bf16)(d2 * inv * gamma2[lane + 64] + beta2[lane + 64]);
  }
  __syncthreads();

  // ---- MLP1 (32x96 @ 96x384) + exact GELU -> g_s (bf16)
  for (int wt = warp; wt < 2 * 24; wt += 8) {
    int tM = wt / 24, tN = wt % 24;
    const __bf16* aP = h_s + (tM * 16 + r) * CH + hf * 8;
    const __bf16* bP = wmlp1T + (size_t)(tN * 16 + r) * CH + hf * 8;
    v8f acc = {};
    for (int k = 0; k < CH; k += 32) {
      bf16x8 a0 = *(const bf16x8*)(aP + k);
      bf16x8 a1 = *(const bf16x8*)(aP + k + 16);
      bf16x8 b0 = *(const bf16x8*)(bP + k);
      bf16x8 b1 = *(const bf16x8*)(bP + k + 16);
      v16bf av = __builtin_shufflevector(a0, a1, 0,1,2,3,4,5,6,7,8,9,10,11,12,13,14,15);
      v16bf bv = __builtin_shufflevector(b0, b1, 0,1,2,3,4,5,6,7,8,9,10,11,12,13,14,15);
      acc = __builtin_amdgcn_wmma_f32_16x16x32_bf16(false, av, false, bv,
                                                    (short)0, acc, false, false);
    }
    int col = tN * 16 + r, rbase = tM * 16 + hf * 8;
    float bc = b_mlp1[col];
    for (int rr = 0; rr < 8; ++rr) {
      int row = rbase + rr;
      float v = acc[rr] + bc;
      float ge = 0.5f * v * (1.0f + erff(v * 0.70710678118654752f));
      g_s[row * (4 * CH) + col] = (__bf16)ge;
    }
  }
  __syncthreads();

  // ---- MLP2 (32x384 @ 384x96) + residual -> out
  for (int wt = warp; wt < 12; wt += 8) {
    int tM = wt / 6, tN = wt % 6;
    const __bf16* aP = g_s + (tM * 16 + r) * (4 * CH) + hf * 8;
    const __bf16* bP = wmlp2T + (size_t)(tN * 16 + r) * (4 * CH) + hf * 8;
    v8f acc = {};
    for (int k = 0; k < 4 * CH; k += 32) {
      bf16x8 a0 = *(const bf16x8*)(aP + k);
      bf16x8 a1 = *(const bf16x8*)(aP + k + 16);
      bf16x8 b0 = *(const bf16x8*)(bP + k);
      bf16x8 b1 = *(const bf16x8*)(bP + k + 16);
      v16bf av = __builtin_shufflevector(a0, a1, 0,1,2,3,4,5,6,7,8,9,10,11,12,13,14,15);
      v16bf bv = __builtin_shufflevector(b0, b1, 0,1,2,3,4,5,6,7,8,9,10,11,12,13,14,15);
      acc = __builtin_amdgcn_wmma_f32_16x16x32_bf16(false, av, false, bv,
                                                    (short)0, acc, false, false);
    }
    int col = tN * 16 + r, rbase = tM * 16 + hf * 8;
    float bc = b_mlp2[col];
    for (int rr = 0; rr < 8; ++rr) {
      int row = rbase + rr;
      out[((size_t)w * MW + row) * CH + col] = acc[rr] + bc + y_s[row * CH + col];
    }
  }
}

// ---------------------------------------------------------------------------
extern "C" void kernel_launch(void* const* d_in, const int* in_sizes, int n_in,
                              void* d_out, int out_size, void* d_ws, size_t ws_size,
                              hipStream_t stream) {
  const float* x      = (const float*)d_in[0];
  const float* coords = (const float*)d_in[1];
  const float* gamma1 = (const float*)d_in[2];
  const float* beta1  = (const float*)d_in[3];
  const float* w_qkv  = (const float*)d_in[4];
  const float* b_qkv  = (const float*)d_in[5];
  const float* t_q    = (const float*)d_in[6];
  const float* t_k    = (const float*)d_in[7];
  const float* t_v    = (const float*)d_in[8];
  const float* w_proj = (const float*)d_in[9];
  const float* b_proj = (const float*)d_in[10];
  const float* gamma2 = (const float*)d_in[11];
  const float* beta2  = (const float*)d_in[12];
  const float* w_mlp1 = (const float*)d_in[13];
  const float* b_mlp1 = (const float*)d_in[14];
  const float* w_mlp2 = (const float*)d_in[15];
  const float* b_mlp2 = (const float*)d_in[16];
  float* out = (float*)d_out;

  char* ws = (char*)d_ws;
  size_t off = 0;
  auto carve = [&](size_t bytes) -> void* {
    void* p = ws + off;
    off = (off + bytes + 255) & ~(size_t)255;
    return p;
  };
  __bf16* wqkvT  = (__bf16*)carve((size_t)96 * 288 * 2);
  __bf16* wprojT = (__bf16*)carve((size_t)96 * 96  * 2);
  __bf16* wmlp1T = (__bf16*)carve((size_t)96 * 384 * 2);
  __bf16* wmlp2T = (__bf16*)carve((size_t)384 * 96 * 2);
  float*  ybuf   = (float*) carve((size_t)NT * CH * 4);

  // 1. weight transposition + bf16 conversion (L2-resident thereafter)
  wprep_kernel<<<(96 * 288 + 255) / 256, 256, 0, stream>>>(w_qkv,  wqkvT,  96, 288);
  wprep_kernel<<<(96 * 96  + 255) / 256, 256, 0, stream>>>(w_proj, wprojT, 96, 96);
  wprep_kernel<<<(96 * 384 + 255) / 256, 256, 0, stream>>>(w_mlp1, wmlp1T, 96, 384);
  wprep_kernel<<<(384 * 96 + 255) / 256, 256, 0, stream>>>(w_mlp2, wmlp2T, 384, 96);

  // 2. fused LN1 + QKV + attention + proj + residual -> y
  constexpr size_t SMEM_A =
      (size_t)(MW * CH * 5 + NH * MW * MW + 4608 + 128) * 4  // f32 arrays
      + (size_t)(MW * CH * 2 + 2 * MW * NH * 3 * 16) * 2     // bf16 arrays
      + (size_t)MW * MW * 2;                                 // idx (u16)
  block_attn_kernel<<<NW, 256, SMEM_A, stream>>>(
      x, coords, gamma1, beta1, wqkvT, b_qkv, t_q, t_k, t_v, wprojT, b_proj, ybuf);

  // 3. fused LN2 + MLP1 + GELU + MLP2 + residual -> out
  constexpr size_t SMEM_B =
      (size_t)MW * CH * 4                  // y_s f32
      + (size_t)MW * CH * 2                // h_s bf16
      + (size_t)MW * 4 * CH * 2;           // g_s bf16
  block_mlp_kernel<<<NW, 256, SMEM_B, stream>>>(
      ybuf, gamma2, beta2, wmlp1T, b_mlp1, wmlp2T, b_mlp2, out);
}
